// SequentialReachingNetwork_88897233092773
// MI455X (gfx1250) — matrix operand order repaired and checked by reading, not compiled, not run on hardware
//
#include <hip/hip_runtime.h>
#include <hip/hip_bf16.h>

#define NHID 2048
#define NBATCH 256
#define DT_C 0.01f
#define DTAU_C 0.2f      // DT/TAU = 0.01/0.05
#define NOISE_C 0.1f

typedef __attribute__((ext_vector_type(16))) __bf16 v16bf;
typedef __attribute__((ext_vector_type(8)))  float  v8f;

// ---------------------------------------------------------------------------
// Feedback MLP 2 -> 10 -> 5 -> N (tanh), evaluated redundantly per thread for
// the tiny hidden layers, then each thread fills 8 of the N outputs.
// ---------------------------------------------------------------------------
__device__ inline void fb_mlp_row(float px, float py,
                                  const float* __restrict__ wf0, const float* __restrict__ bf0,
                                  const float* __restrict__ wf1, const float* __restrict__ bf1,
                                  const float* __restrict__ wf2, const float* __restrict__ bf2,
                                  float* __restrict__ fbrow, int tid) {
    float h1[10];
#pragma unroll
    for (int j = 0; j < 10; ++j)
        h1[j] = tanhf(px * wf0[j] + py * wf0[10 + j] + bf0[j]);
    float h2[5];
#pragma unroll
    for (int k = 0; k < 5; ++k) {
        float s = bf1[k];
#pragma unroll
        for (int j = 0; j < 10; ++j) s += h1[j] * wf1[j * 5 + k];
        h2[k] = tanhf(s);
    }
    for (int n = tid; n < NHID; n += 256) {
        float s = bf2[n];
#pragma unroll
        for (int k = 0; k < 5; ++k) s += h2[k] * wf2[k * NHID + n];
        fbrow[n] = tanhf(s);
    }
}

// ---------------------------------------------------------------------------
// Init: static drive u0 = targMLP(target_cfg) + trigMLP(trigger) + B.  One
// block per batch row.
// ---------------------------------------------------------------------------
__global__ void srn_init_u0(const float* __restrict__ wt0, const float* __restrict__ bt0,
                            const float* __restrict__ wt1, const float* __restrict__ bt1,
                            const float* __restrict__ wt2, const float* __restrict__ bt2,
                            const float* __restrict__ wt3, const float* __restrict__ bt3,
                            const float* __restrict__ wg0, const float* __restrict__ bg0,
                            const float* __restrict__ wg1, const float* __restrict__ bg1,
                            const float* __restrict__ tcfg, const float* __restrict__ trig,
                            const float* __restrict__ Bvec, float* __restrict__ u0) {
    __shared__ float h1[50], h2[25], h3[12], g1s;
    const int b = blockIdx.x, tid = threadIdx.x;
    if (tid < 50) {
        float s = bt0[tid];
#pragma unroll
        for (int i = 0; i < 6; ++i) s += tcfg[b * 6 + i] * wt0[i * 50 + tid];
        h1[tid] = tanhf(s);
    }
    __syncthreads();
    if (tid < 25) {
        float s = bt1[tid];
        for (int j = 0; j < 50; ++j) s += h1[j] * wt1[j * 25 + tid];
        h2[tid] = tanhf(s);
    }
    __syncthreads();
    if (tid < 12) {
        float s = bt2[tid];
        for (int j = 0; j < 25; ++j) s += h2[j] * wt2[j * 12 + tid];
        h3[tid] = tanhf(s);
    }
    if (tid == 0) g1s = tanhf(trig[b] * wg0[0] + bg0[0]);
    __syncthreads();
    const float g1 = g1s;
    for (int n = tid; n < NHID; n += 256) {
        float s = bt3[n];
#pragma unroll
        for (int k = 0; k < 12; ++k) s += h3[k] * wt3[k * NHID + n];
        u0[(size_t)b * NHID + n] = tanhf(s) + tanhf(g1 * wg1[n] + bg1[n]) + Bvec[n];
    }
}

// ---------------------------------------------------------------------------
// Init: transpose + convert J (fp32 row-major [k][n]) -> Jt bf16 [n][k].
// ---------------------------------------------------------------------------
__global__ void srn_transpose_J(const float* __restrict__ J, __hip_bfloat16* __restrict__ Jt) {
    __shared__ float tile[32][33];
    const int colBase = blockIdx.x * 32, rowBase = blockIdx.y * 32;
    const int tx = threadIdx.x, ty = threadIdx.y;
#pragma unroll
    for (int j = 0; j < 4; ++j) {
        int r = ty + j * 8;
        tile[r][tx] = J[(size_t)(rowBase + r) * NHID + colBase + tx];
    }
    __syncthreads();
#pragma unroll
    for (int j = 0; j < 4; ++j) {
        int r = ty + j * 8;
        Jt[(size_t)(colBase + r) * NHID + rowBase + tx] = __float2bfloat16(tile[tx][r]);
    }
}

// ---------------------------------------------------------------------------
// Init: x = x0, r0 = tanh(x0) (bf16), pos = pos0, fb0 = MLP(pos0).
// ---------------------------------------------------------------------------
__global__ void srn_init_state(const float* __restrict__ x0, const float* __restrict__ pos0,
                               const float* __restrict__ wf0, const float* __restrict__ bf0,
                               const float* __restrict__ wf1, const float* __restrict__ bf1,
                               const float* __restrict__ wf2, const float* __restrict__ bf2,
                               float* __restrict__ xbuf, __hip_bfloat16* __restrict__ r0,
                               float* __restrict__ pos, float* __restrict__ fbbuf) {
    const int b = blockIdx.x, tid = threadIdx.x;
    if (tid < 2) pos[b * 2 + tid] = pos0[b * 2 + tid];
    for (int n = tid; n < NHID; n += 256) {
        size_t idx = (size_t)b * NHID + n;
        float x = x0[idx];
        xbuf[idx] = x;
        r0[idx] = __float2bfloat16(tanhf(x));
    }
    fb_mlp_row(pos0[b * 2], pos0[b * 2 + 1], wf0, bf0, wf1, bf1, wf2, bf2,
               fbbuf + (size_t)b * NHID, tid);
}

// ---------------------------------------------------------------------------
// Step GEMM + state update.  256 threads = 8 waves.  Each wave owns a
// 16(m) x 32(n) output tile: the A fragment (r tile) is loaded once per
// k-chunk and reused for two WMMAs (3 b128 loads per WMMA instead of 4).
// WG tile = 32 (batch) x 128 (hidden); grid = 16 x 8 = 128 WGs.
// Epilogue: x_new = x + dtau*(-x + u0 + fb + y + 0.1*noise),
//           r_next = tanh(x_old)  (quirk: tanh of PRE-update state).
// ---------------------------------------------------------------------------
__global__ void __launch_bounds__(256)
srn_step(const __hip_bfloat16* __restrict__ r_in, __hip_bfloat16* __restrict__ r_out,
         float* __restrict__ xbuf, const __hip_bfloat16* __restrict__ Jt,
         const float* __restrict__ u0, const float* __restrict__ fb,
         const float* __restrict__ noise_t) {
    const int tid  = threadIdx.x;
    const int wave = tid >> 5;
    const int lane = tid & 31;
    const int half = lane >> 4;   // lane group 0/1
    const int lm   = lane & 15;

    const int mBase = blockIdx.y * 32  + (wave & 1) * 16;   // batch tile (16 rows)
    const int nTile = blockIdx.x * 128 + (wave >> 1) * 32;  // hidden tile pair (32 cols)

    const __hip_bfloat16* __restrict__ arow  = r_in + (size_t)(mBase + lm) * NHID;
    const __hip_bfloat16* __restrict__ brow0 = Jt + (size_t)(nTile + lm) * NHID;
    const __hip_bfloat16* __restrict__ brow1 = Jt + (size_t)(nTile + 16 + lm) * NHID;

    union Frag { v16bf v; uint4 q[2]; };
    v8f c0 = {};
    v8f c1 = {};

#pragma unroll 2
    for (int k0 = 0; k0 < NHID; k0 += 32) {
        Frag A, B0, B1;
        // A 16x32 bf16: element e -> k = e + 16*half
        A.q[0] = *(const uint4*)(arow + k0 + 16 * half);
        A.q[1] = *(const uint4*)(arow + k0 + 16 * half + 8);
        // B 32x16 bf16 (column n per lane, Jt is [n][k]): element e -> k = e + 16*half
        B0.q[0] = *(const uint4*)(brow0 + k0 + 16 * half);
        B0.q[1] = *(const uint4*)(brow0 + k0 + 16 * half + 8);
        B1.q[0] = *(const uint4*)(brow1 + k0 + 16 * half);
        B1.q[1] = *(const uint4*)(brow1 + k0 + 16 * half + 8);
        if (k0 + 32 < NHID) {
            __builtin_prefetch(brow0 + k0 + 32, 0, 3);
            __builtin_prefetch(brow1 + k0 + 32, 0, 3);
            __builtin_prefetch(arow + k0 + 32, 0, 3);
        }
        c0 = __builtin_amdgcn_wmma_f32_16x16x32_bf16(
                 false, A.v, false, B0.v, (short)0, c0, false, false);
        c1 = __builtin_amdgcn_wmma_f32_16x16x32_bf16(
                 false, A.v, false, B1.v, (short)0, c1, false, false);
    }

    // Epilogue: C layout lane L, vgpr v -> (m = v + 8*half, n = lane&15)
    const int n0 = nTile + lm;
    const int n1 = nTile + 16 + lm;
#pragma unroll
    for (int v = 0; v < 8; ++v) {
        const int m = mBase + v + 8 * half;
        const size_t row = (size_t)m * NHID;
        {
            const size_t idx = row + n0;
            float xo = xbuf[idx];
            float xn = xo + DTAU_C * (-xo + u0[idx] + fb[idx] + c0[v] + NOISE_C * noise_t[idx]);
            xbuf[idx] = xn;
            r_out[idx] = __float2bfloat16(tanhf(xo));
        }
        {
            const size_t idx = row + n1;
            float xo = xbuf[idx];
            float xn = xo + DTAU_C * (-xo + u0[idx] + fb[idx] + c1[v] + NOISE_C * noise_t[idx]);
            xbuf[idx] = xn;
            r_out[idx] = __float2bfloat16(tanhf(xo));
        }
    }
}

// ---------------------------------------------------------------------------
// Post-step: pos += DT * (r_new @ Wout)  (deterministic tree reduction),
// write pos to out[t], compute fb_{t+1} = MLP(pos) for the next step.
// One block per batch row.
// ---------------------------------------------------------------------------
__global__ void __launch_bounds__(256)
srn_post(const __hip_bfloat16* __restrict__ r_new, const float* __restrict__ Wout,
         float* __restrict__ pos, float* __restrict__ out_t,
         const float* __restrict__ wf0, const float* __restrict__ bf0,
         const float* __restrict__ wf1, const float* __restrict__ bf1,
         const float* __restrict__ wf2, const float* __restrict__ bf2,
         float* __restrict__ fbbuf) {
    __shared__ float red0[256], red1[256], posS[2];
    const int b = blockIdx.x, tid = threadIdx.x;
    float s0 = 0.f, s1 = 0.f;
    for (int n = tid; n < NHID; n += 256) {
        float rv = __bfloat162float(r_new[(size_t)b * NHID + n]);
        s0 += rv * Wout[2 * n];
        s1 += rv * Wout[2 * n + 1];
    }
    red0[tid] = s0; red1[tid] = s1;
    __syncthreads();
    for (int s = 128; s > 0; s >>= 1) {
        if (tid < s) { red0[tid] += red0[tid + s]; red1[tid] += red1[tid + s]; }
        __syncthreads();
    }
    if (tid == 0) {
        float p0 = pos[2 * b]     + DT_C * red0[0];
        float p1 = pos[2 * b + 1] + DT_C * red1[0];
        pos[2 * b] = p0; pos[2 * b + 1] = p1;
        out_t[2 * b] = p0; out_t[2 * b + 1] = p1;
        posS[0] = p0; posS[1] = p1;
    }
    __syncthreads();
    fb_mlp_row(posS[0], posS[1], wf0, bf0, wf1, bf1, wf2, bf2,
               fbbuf + (size_t)b * NHID, tid);
}

// ---------------------------------------------------------------------------
extern "C" void kernel_launch(void* const* d_in, const int* in_sizes, int n_in,
                              void* d_out, int out_size, void* d_ws, size_t ws_size,
                              hipStream_t stream) {
    const float* wt0 = (const float*)d_in[0];  const float* bt0 = (const float*)d_in[1];
    const float* wt1 = (const float*)d_in[2];  const float* bt1 = (const float*)d_in[3];
    const float* wt2 = (const float*)d_in[4];  const float* bt2 = (const float*)d_in[5];
    const float* wt3 = (const float*)d_in[6];  const float* bt3 = (const float*)d_in[7];
    const float* wf0 = (const float*)d_in[8];  const float* bf0 = (const float*)d_in[9];
    const float* wf1 = (const float*)d_in[10]; const float* bf1 = (const float*)d_in[11];
    const float* wf2 = (const float*)d_in[12]; const float* bf2 = (const float*)d_in[13];
    const float* wg0 = (const float*)d_in[14]; const float* bg0 = (const float*)d_in[15];
    const float* wg1 = (const float*)d_in[16]; const float* bg1 = (const float*)d_in[17];
    const float* J    = (const float*)d_in[18];
    const float* Bvec = (const float*)d_in[19];
    const float* Wout = (const float*)d_in[20];
    const float* x0   = (const float*)d_in[21];
    const float* pos0 = (const float*)d_in[22];
    const float* tcfg = (const float*)d_in[23];
    const float* trig = (const float*)d_in[24];
    const float* noise = (const float*)d_in[25];
    float* out = (float*)d_out;

    const int T = in_sizes[25] / (NBATCH * NHID);   // 300

    // ---- workspace layout ----
    char* base = (char*)d_ws;
    size_t off = 0;
    __hip_bfloat16* Jt = (__hip_bfloat16*)(base + off); off += (size_t)NHID * NHID * 2;      // 8 MB
    float* u0buf = (float*)(base + off);               off += (size_t)NBATCH * NHID * 4;     // 2 MB
    float* xbuf  = (float*)(base + off);               off += (size_t)NBATCH * NHID * 4;     // 2 MB
    float* fbbuf = (float*)(base + off);               off += (size_t)NBATCH * NHID * 4;     // 2 MB
    __hip_bfloat16* rb0 = (__hip_bfloat16*)(base + off); off += (size_t)NBATCH * NHID * 2;   // 1 MB
    __hip_bfloat16* rb1 = (__hip_bfloat16*)(base + off); off += (size_t)NBATCH * NHID * 2;   // 1 MB
    float* posbuf = (float*)(base + off);              off += (size_t)NBATCH * 2 * 4;

    // ---- one-time per call initialization ----
    srn_transpose_J<<<dim3(NHID / 32, NHID / 32), dim3(32, 8), 0, stream>>>(J, Jt);
    srn_init_u0<<<NBATCH, 256, 0, stream>>>(wt0, bt0, wt1, bt1, wt2, bt2, wt3, bt3,
                                            wg0, bg0, wg1, bg1, tcfg, trig, Bvec, u0buf);
    srn_init_state<<<NBATCH, 256, 0, stream>>>(x0, pos0, wf0, bf0, wf1, bf1, wf2, bf2,
                                               xbuf, rb0, posbuf, fbbuf);

    // ---- sequential RNN ----
    for (int t = 0; t < T; ++t) {
        __hip_bfloat16* rin  = (t & 1) ? rb1 : rb0;
        __hip_bfloat16* rout = (t & 1) ? rb0 : rb1;
        srn_step<<<dim3(NHID / 128, NBATCH / 32), 256, 0, stream>>>(
            rin, rout, xbuf, Jt, u0buf, fbbuf, noise + (size_t)t * NBATCH * NHID);
        srn_post<<<NBATCH, 256, 0, stream>>>(
            rout, Wout, posbuf, out + (size_t)t * NBATCH * 2,
            wf0, bf0, wf1, bf1, wf2, bf2, fbbuf);
    }
    (void)n_in; (void)out_size; (void)ws_size;
}
